// DisAwareExpertChoiceMoE_23691039604950
// MI455X (gfx1250) — matrix operand order repaired and checked by reading, not compiled, not run on hardware
//
#include <hip/hip_runtime.h>
#include <hip/hip_bf16.h>
#include <math.h>

// ---------------- problem constants ----------------
constexpr int NTOK = 65536;
constexpr int CD = 192;   // input size
constexpr int OD = 192;   // output size
constexpr int HD = 384;   // hidden size
constexpr int ED = 8;     // experts
constexpr int BD = 4;     // bands
constexpr int RD = 8;     // lora rank
constexpr int FD = 64;    // dct features
constexpr int GIN = CD + FD + 2;      // 258
constexpr int KCAP = 10240;           // expert capacity
constexpr float LORA_SCALE = 2.0f;    // ALPHA / R = 16/8
constexpr int TM = 32;                // tokens per expert block (2 WMMA M-tiles)

typedef __attribute__((ext_vector_type(16))) _Float16 v16h;
typedef __attribute__((ext_vector_type(8)))  _Float16 v8h;
typedef __attribute__((ext_vector_type(8)))  float    v8f;

// ---------------- helpers ----------------
__device__ __forceinline__ unsigned fkey(float f) {
  unsigned u = __float_as_uint(f);
  return (u & 0x80000000u) ? ~u : (u | 0x80000000u);
}

__device__ __forceinline__ v8f wmma16(v16h a, v16h b, v8f c) {
  // v_wmma_f32_16x16x32_f16 : D = A(16x32 f16) x B(32x16 f16) + C(f32)
  return __builtin_amdgcn_wmma_f32_16x16x32_f16(false, a, false, b, (short)0, c,
                                                false, false);
}

// A-fragment: lane holds row M=lane%16; halfs[0:8)=K kpos..kpos+7, halfs[8:16)=K kpos+16..kpos+23
__device__ __forceinline__ v16h load_a(const _Float16* rowp, int kpos) {
  v8h lo = *(const v8h*)(rowp + kpos);
  v8h hi = *(const v8h*)(rowp + kpos + 16);
  v16h a;
#pragma unroll
  for (int i = 0; i < 8; ++i) { a[i] = lo[i]; a[i + 8] = hi[i]; }
  return a;
}

// B-fragment: lane holds column N=lane%16; halfs[0:8)=K p..p+7, halfs[8:16)=K p+16..p+23
__device__ __forceinline__ v16h load_b(const _Float16* p) {
  v8h lo = *(const v8h*)(p);
  v8h hi = *(const v8h*)(p + 16);
  v16h b;
#pragma unroll
  for (int i = 0; i < 8; ++i) { b[i] = lo[i]; b[i + 8] = hi[i]; }
  return b;
}

// ---------------- kernel 0: init ----------------
__global__ __launch_bounds__(256) void init_kernel(float* __restrict__ out,
                                                   int* __restrict__ cnt,
                                                   int nout) {
  int g = blockIdx.x * blockDim.x + threadIdx.x;
  int stride = gridDim.x * blockDim.x;
  for (int i = g; i < nout; i += stride) out[i] = 0.f;
  if (g < ED) cnt[g] = 0;
}

// ---------------- kernel 1: weight conversion to f16 transposed ----------------
__global__ __launch_bounds__(256) void prep_kernel(
    const float* __restrict__ W1, const float* __restrict__ W2,
    const float* __restrict__ A1, const float* __restrict__ B1,
    const float* __restrict__ A2, const float* __restrict__ B2,
    _Float16* __restrict__ w1t, _Float16* __restrict__ w2t,
    _Float16* __restrict__ a1t, _Float16* __restrict__ b1t,
    _Float16* __restrict__ a2t, _Float16* __restrict__ b2t) {
  int g = blockIdx.x * blockDim.x + threadIdx.x;
  int stride = gridDim.x * blockDim.x;
  // w1t[(e*HD+h)*CD + c] = W1[e][c][h]
  for (int i = g; i < ED * HD * CD; i += stride) {
    int e = i / (HD * CD); int r = i % (HD * CD); int h = r / CD; int c = r % CD;
    w1t[i] = (_Float16)W1[((size_t)(e * CD + c)) * HD + h];
  }
  // w2t[(e*OD+o)*HD + hk] = W2[e][hk][o]
  for (int i = g; i < ED * OD * HD; i += stride) {
    int e = i / (OD * HD); int r = i % (OD * HD); int o = r / HD; int hk = r % HD;
    w2t[i] = (_Float16)W2[((size_t)(e * HD + hk)) * OD + o];
  }
  // a1t[(e*32+j)*CD + c] = A1[e][b][c][r], j=b*8+r
  for (int i = g; i < ED * 32 * CD; i += stride) {
    int e = i / (32 * CD); int r2 = i % (32 * CD); int j = r2 / CD; int c = r2 % CD;
    int b = j >> 3, rr = j & 7;
    a1t[i] = (_Float16)A1[(((size_t)(e * BD + b) * CD + c) * RD) + rr];
  }
  // b1t[(e*HD+h)*32 + j] = B1[e][b][r][h]
  for (int i = g; i < ED * HD * 32; i += stride) {
    int e = i / (HD * 32); int r2 = i % (HD * 32); int h = r2 / 32; int j = r2 % 32;
    int b = j >> 3, rr = j & 7;
    b1t[i] = (_Float16)B1[(((size_t)(e * BD + b) * RD + rr) * HD) + h];
  }
  // a2t[(e*32+j)*HD + hk] = A2[e][b][hk][r]
  for (int i = g; i < ED * 32 * HD; i += stride) {
    int e = i / (32 * HD); int r2 = i % (32 * HD); int j = r2 / HD; int hk = r2 % HD;
    int b = j >> 3, rr = j & 7;
    a2t[i] = (_Float16)A2[(((size_t)(e * BD + b) * HD + hk) * RD) + rr];
  }
  // b2t[(e*OD+o)*32 + j] = B2[e][b][r][o]
  for (int i = g; i < ED * OD * 32; i += stride) {
    int e = i / (OD * 32); int r2 = i % (OD * 32); int o = r2 / 32; int j = r2 % 32;
    int b = j >> 3, rr = j & 7;
    b2t[i] = (_Float16)B2[(((size_t)(e * BD + b) * RD + rr) * OD) + o];
  }
}

// ---------------- kernel 2: gating logits (fp32) ----------------
__global__ __launch_bounds__(256) void gate_kernel(
    const float* __restrict__ x, const float* __restrict__ xp,
    const float* __restrict__ Wz, const float* __restrict__ lng,
    const float* __restrict__ lnb, const float* __restrict__ Wg,
    const float* __restrict__ bg, float* __restrict__ logits) {
  int n = blockIdx.x * 256 + threadIdx.x;
  if (n >= NTOK) return;
  const float* xr = x + (size_t)n * CD;
  const float* xpr = xp + (size_t)n * CD;
  // residual hint stats (unbiased std like torch)
  float s1 = 0.f, s2 = 0.f;
  for (int c = 0; c < CD; ++c) {
    float a = fabsf(xr[c] - xpr[c]);
    s1 += a; s2 += a * a;
  }
  float mean = s1 / (float)CD;
  float var = fmaxf((s2 - (float)CD * mean * mean) / (float)(CD - 1), 0.f);
  float mu = log1pf(mean);
  float sd = log1pf(sqrtf(var));
  // Z = x @ Wz
  float Z[FD];
#pragma unroll
  for (int f = 0; f < FD; ++f) Z[f] = 0.f;
  for (int c = 0; c < CD; ++c) {
    float xv = xr[c];
    const float* wr = Wz + (size_t)c * FD;
#pragma unroll
    for (int f = 0; f < FD; ++f) Z[f] = fmaf(xv, wr[f], Z[f]);
  }
  // LayerNorm stats over 258-dim et = [x, Z, mu, sd]
  float sm = 0.f, sq = 0.f;
  for (int c = 0; c < CD; ++c) { float v = xr[c]; sm += v; sq += v * v; }
#pragma unroll
  for (int f = 0; f < FD; ++f) { float v = Z[f]; sm += v; sq += v * v; }
  sm += mu + sd; sq += mu * mu + sd * sd;
  float m = sm / (float)GIN;
  float v2 = fmaxf(sq / (float)GIN - m * m, 0.f);
  float r = rsqrtf(v2 + 1e-5f);
  // logits = LN(et) @ Wg + bg
  float acc[ED];
#pragma unroll
  for (int e = 0; e < ED; ++e) acc[e] = bg[e];
  for (int c = 0; c < CD; ++c) {
    float vn = (xr[c] - m) * r * lng[c] + lnb[c];
    const float* wg = Wg + (size_t)c * ED;
#pragma unroll
    for (int e = 0; e < ED; ++e) acc[e] = fmaf(vn, wg[e], acc[e]);
  }
#pragma unroll
  for (int f = 0; f < FD; ++f) {
    int j = CD + f;
    float vn = (Z[f] - m) * r * lng[j] + lnb[j];
    const float* wg = Wg + (size_t)j * ED;
#pragma unroll
    for (int e = 0; e < ED; ++e) acc[e] = fmaf(vn, wg[e], acc[e]);
  }
  {
    int j = CD + FD;
    float vn = (mu - m) * r * lng[j] + lnb[j];
#pragma unroll
    for (int e = 0; e < ED; ++e) acc[e] = fmaf(vn, Wg[(size_t)j * ED + e], acc[e]);
    j = CD + FD + 1;
    vn = (sd - m) * r * lng[j] + lnb[j];
#pragma unroll
    for (int e = 0; e < ED; ++e) acc[e] = fmaf(vn, Wg[(size_t)j * ED + e], acc[e]);
  }
#pragma unroll
  for (int e = 0; e < ED; ++e) logits[(size_t)n * ED + e] = acc[e];
}

// ---------------- kernel 3: per-expert K-th largest via radix select ----------------
__global__ __launch_bounds__(256) void topk_kernel(const float* __restrict__ logits,
                                                   unsigned* __restrict__ thr) {
  __shared__ unsigned hist[2048];
  __shared__ unsigned s_prefix;
  __shared__ int s_rem;
  const int e = blockIdx.x;
  const int tid = threadIdx.x;
  if (tid == 0) { s_prefix = 0u; s_rem = KCAP; }
  const int shifts[3] = {21, 10, 0};
  const int nbits[3] = {11, 11, 10};
  for (int p = 0; p < 3; ++p) {
    for (int i = tid; i < 2048; i += 256) hist[i] = 0u;
    __syncthreads();
    const unsigned pref = s_prefix;
    const int sh = shifts[p], nb = nbits[p];
    const unsigned msk = (1u << nb) - 1u;
    for (int n = tid; n < NTOK; n += 256) {
      unsigned u = fkey(logits[(size_t)n * ED + e]);
      bool ok = (p == 0) || ((u >> (sh + nb)) == pref);
      if (ok) atomicAdd(&hist[(u >> sh) & msk], 1u);
    }
    __syncthreads();
    if (tid == 0) {
      int rem = s_rem;
      unsigned pr = s_prefix;
      for (int b = (int)msk; b >= 0; --b) {
        int c = (int)hist[b];
        if (rem > c) rem -= c;
        else { pr = (pr << nb) | (unsigned)b; break; }
      }
      s_rem = rem; s_prefix = pr;
    }
    __syncthreads();
  }
  if (tid == 0) thr[e] = s_prefix;  // key of K-th largest logit for this expert
}

// ---------------- kernel 4: coverage + top-2 softmax + dispatch ----------------
__global__ __launch_bounds__(256) void route_kernel(
    const float* __restrict__ logits, const unsigned* __restrict__ thr,
    int* __restrict__ tidx, float* __restrict__ tw, int* __restrict__ cnt,
    int* __restrict__ ltok, float* __restrict__ lwv) {
  int n = blockIdx.x * 256 + threadIdx.x;
  if (n >= NTOK) return;
  float lg[ED]; bool mk[ED]; bool any = false;
#pragma unroll
  for (int e = 0; e < ED; ++e) {
    lg[e] = logits[(size_t)n * ED + e];
    mk[e] = fkey(lg[e]) >= thr[e];
    any = any || mk[e];
  }
  if (!any) {
    int best = 0; float bv = lg[0];
#pragma unroll
    for (int e = 1; e < ED; ++e) if (lg[e] > bv) { bv = lg[e]; best = e; }
#pragma unroll
    for (int e = 0; e < ED; ++e) mk[e] = mk[e] || (e == best);
  }
  int e0 = -1; float v0 = -3.4e38f;
#pragma unroll
  for (int e = 0; e < ED; ++e) if (mk[e] && lg[e] > v0) { v0 = lg[e]; e0 = e; }
  int e1 = -1; float v1 = -3.4e38f;
#pragma unroll
  for (int e = 0; e < ED; ++e) if (mk[e] && e != e0 && lg[e] > v1) { v1 = lg[e]; e1 = e; }
  float w0 = 1.f, w1 = 0.f;
  if (e1 >= 0) {
    float d = expf(v1 - v0);
    float s = 1.f + d;
    w0 = 1.f / s; w1 = d / s;
  }
  tidx[2 * n] = e0;            tw[2 * n] = w0;
  tidx[2 * n + 1] = e1;        tw[2 * n + 1] = w1;
  int p = atomicAdd(&cnt[e0], 1);
  ltok[(size_t)e0 * NTOK + p] = n; lwv[(size_t)e0 * NTOK + p] = w0;
  if (e1 >= 0 && w1 > 0.f) {
    int q = atomicAdd(&cnt[e1], 1);
    ltok[(size_t)e1 * NTOK + q] = n; lwv[(size_t)e1 * NTOK + q] = w1;
  }
}

// ---------------- kernel 5: WMMA expert MLP + LoRA ----------------
#define XS 200  // xt row stride (halfs): 100 dwords
#define HS 392  // ht row stride (halfs): 196 dwords

__global__ __launch_bounds__(128, 1) void expert_kernel(
    const float* __restrict__ x, const float* __restrict__ bwg,
    const float* __restrict__ b1, const float* __restrict__ b2,
    const _Float16* __restrict__ w1t, const _Float16* __restrict__ w2t,
    const _Float16* __restrict__ a1t, const _Float16* __restrict__ b1t,
    const _Float16* __restrict__ a2t, const _Float16* __restrict__ b2t,
    const int* __restrict__ counts, const int* __restrict__ ltok,
    const float* __restrict__ lwv, float* __restrict__ out) {
  const int e = blockIdx.y;
  const int cnt = counts[e];
  const int base = blockIdx.x * TM;
  if (base >= cnt) return;
  const int rows = min(TM, cnt - base);

  __shared__ _Float16 xt[TM][XS];
  __shared__ _Float16 ht[TM][HS];
  __shared__ _Float16 t1b[TM][32];
  __shared__ _Float16 t2b[TM][32];
  __shared__ float bwt[TM][BD];
  __shared__ float gwt[TM];
  __shared__ int tokid[TM];

  const int tid = threadIdx.x;
  const int wv = tid >> 5;
  const int lane = tid & 31;
  const int col = lane & 15;
  const int asel = (lane < 16) ? 0 : 8;    // A-fragment / C-row offset
  const int bsel = (lane < 16) ? 0 : 8;    // B-fragment K offset

  if (tid < TM) {
    int m = tid; int t = -1; float w = 0.f;
    if (m < rows) {
      t = ltok[(size_t)e * NTOK + base + m];
      w = lwv[(size_t)e * NTOK + base + m];
    }
    tokid[m] = t; gwt[m] = w;
    for (int b = 0; b < BD; ++b) bwt[m][b] = (t >= 0) ? bwg[(size_t)t * BD + b] : 0.f;
  }
  __syncthreads();
  for (int i = tid; i < TM * CD; i += 128) {
    int m = i / CD, c = i % CD;
    int t = tokid[m];
    xt[m][c] = (t >= 0) ? (_Float16)x[(size_t)t * CD + c] : (_Float16)0.f;
  }
  __syncthreads();

  // ---- stage 1: t1b = (scale * bw) .* (x @ A1t); 32x32 = 4 tiles, one per wave
  //      split-K into two independent WMMA chains to hide the RAW hazard
  {
    const int mh = (wv & 1) * 16;
    const int jb = (wv >> 1) * 16;
    const _Float16* bp = a1t + ((size_t)(e * 32 + jb + col)) * CD;
    v8f accA = {}, accB = {};
#pragma unroll
    for (int k = 0; k < CD / 2; k += 32) {
      v16h a0 = load_a(&xt[mh + col][0], k + asel);
      v16h b0 = load_b(bp + k + bsel);
      v16h a1 = load_a(&xt[mh + col][0], (CD / 2) + k + asel);
      v16h b1f = load_b(bp + (CD / 2) + k + bsel);
      accA = wmma16(a0, b0, accA);
      accB = wmma16(a1, b1f, accB);
    }
#pragma unroll
    for (int i = 0; i < 8; ++i) {
      int m = mh + i + asel;
      int j = jb + col;
      float f = LORA_SCALE * bwt[m][j >> 3];
      t1b[m][j] = (_Float16)((accA[i] + accB[i]) * f);
    }
  }
  __syncthreads();

  // ---- stage 2: h = gelu(x@W1 + b1 + t1b@B1t); 6 hidden cols x 2 M-tiles per wave
#pragma unroll 1
  for (int t6 = 0; t6 < 6; ++t6) {
    const int h = (wv * 6 + t6) * 16 + col;
    const _Float16* bp = w1t + ((size_t)(e * HD + h)) * CD;
    v8f acc0 = {}, acc1 = {};
#pragma unroll
    for (int k = 0; k < CD; k += 32) {
      v16h b = load_b(bp + k + bsel);
      v16h a0 = load_a(&xt[col][0], k + asel);
      v16h a1 = load_a(&xt[16 + col][0], k + asel);
      acc0 = wmma16(a0, b, acc0);   // two independent chains share one B fragment:
      acc1 = wmma16(a1, b, acc1);   // hides WMMA->WMMA RAW hazard, halves B traffic
    }
    v8f u0 = {}, u1v = {};
    {
      v16h b = load_b(b1t + ((size_t)(e * HD + h)) * 32 + bsel);
      v16h a0 = load_a(&t1b[col][0], asel);
      v16h a1 = load_a(&t1b[16 + col][0], asel);
      u0 = wmma16(a0, b, u0);
      u1v = wmma16(a1, b, u1v);
    }
    const float bias = b1[e * HD + h];
#pragma unroll
    for (int i = 0; i < 8; ++i) {
      float v = acc0[i] + u0[i] + bias;
      ht[i + asel][h] = (_Float16)(0.5f * v * (1.0f + erff(v * 0.70710678118654752f)));
      float v2 = acc1[i] + u1v[i] + bias;
      ht[16 + i + asel][h] = (_Float16)(0.5f * v2 * (1.0f + erff(v2 * 0.70710678118654752f)));
    }
  }
  __syncthreads();

  // ---- stage 3: t2b = (scale * bw) .* (h @ A2t); 4 tiles, split-K dual chains
  {
    const int mh = (wv & 1) * 16;
    const int jb = (wv >> 1) * 16;
    const _Float16* bp = a2t + ((size_t)(e * 32 + jb + col)) * HD;
    v8f accA = {}, accB = {};
#pragma unroll
    for (int k = 0; k < HD / 2; k += 32) {
      v16h a0 = load_a(&ht[mh + col][0], k + asel);
      v16h b0 = load_b(bp + k + bsel);
      v16h a1 = load_a(&ht[mh + col][0], (HD / 2) + k + asel);
      v16h b1f = load_b(bp + (HD / 2) + k + bsel);
      accA = wmma16(a0, b0, accA);
      accB = wmma16(a1, b1f, accB);
    }
#pragma unroll
    for (int i = 0; i < 8; ++i) {
      int m = mh + i + asel;
      int j = jb + col;
      float f = LORA_SCALE * bwt[m][j >> 3];
      t2b[m][j] = (_Float16)((accA[i] + accB[i]) * f);
    }
  }
  __syncthreads();

  // ---- stage 4: y = h@W2 + b2 + t2b@B2t ; out[token] += gate * y
#pragma unroll 1
  for (int t3 = 0; t3 < 3; ++t3) {
    const int o = (wv * 3 + t3) * 16 + col;
    const _Float16* bp = w2t + ((size_t)(e * OD + o)) * HD;
    v8f acc0 = {}, acc1 = {};
#pragma unroll
    for (int k = 0; k < HD; k += 32) {
      v16h b = load_b(bp + k + bsel);
      v16h a0 = load_a(&ht[col][0], k + asel);
      v16h a1 = load_a(&ht[16 + col][0], k + asel);
      acc0 = wmma16(a0, b, acc0);
      acc1 = wmma16(a1, b, acc1);
    }
    v8f u0 = {}, u1v = {};
    {
      v16h b = load_b(b2t + ((size_t)(e * OD + o)) * 32 + bsel);
      v16h a0 = load_a(&t2b[col][0], asel);
      v16h a1 = load_a(&t2b[16 + col][0], asel);
      u0 = wmma16(a0, b, u0);
      u1v = wmma16(a1, b, u1v);
    }
    const float bias = b2[e * OD + o];
#pragma unroll
    for (int i = 0; i < 8; ++i) {
      int m0 = i + asel;
      float w0 = gwt[m0];
      if (w0 != 0.f)
        atomicAdd(&out[(size_t)tokid[m0] * OD + o], (acc0[i] + u0[i] + bias) * w0);
      int m1 = 16 + i + asel;
      float w1 = gwt[m1];
      if (w1 != 0.f)
        atomicAdd(&out[(size_t)tokid[m1] * OD + o], (acc1[i] + u1v[i] + bias) * w1);
    }
  }
}

// ---------------- kernel 6: aux loss (deterministic reduction) ----------------
__global__ __launch_bounds__(256) void loss_kernel(const int* __restrict__ tidx,
                                                   const float* __restrict__ tw,
                                                   float* __restrict__ out_loss) {
  __shared__ float red[256];
  float imp[ED], ld[ED];
#pragma unroll
  for (int e = 0; e < ED; ++e) { imp[e] = 0.f; ld[e] = 0.f; }
  for (int n = threadIdx.x; n < NTOK; n += 256) {
#pragma unroll
    for (int s = 0; s < 2; ++s) {
      int e = tidx[2 * n + s];
      float w = tw[2 * n + s];
#pragma unroll
      for (int q = 0; q < ED; ++q) {
        if (q == e) { imp[q] += w; if (w > 0.f) ld[q] += 1.f; }
      }
    }
  }
  float impT[ED], ldT[ED];
  for (int e = 0; e < ED; ++e) {
    red[threadIdx.x] = imp[e]; __syncthreads();
    for (int s = 128; s > 0; s >>= 1) {
      if (threadIdx.x < s) red[threadIdx.x] += red[threadIdx.x + s];
      __syncthreads();
    }
    impT[e] = red[0]; __syncthreads();
    red[threadIdx.x] = ld[e]; __syncthreads();
    for (int s = 128; s > 0; s >>= 1) {
      if (threadIdx.x < s) red[threadIdx.x] += red[threadIdx.x + s];
      __syncthreads();
    }
    ldT[e] = red[0]; __syncthreads();
  }
  if (threadIdx.x == 0) {
    float loss = 0.f;
    {
      float m = 0.f;
      for (int e = 0; e < ED; ++e) m += impT[e];
      m /= (float)ED;
      float var = 0.f;
      for (int e = 0; e < ED; ++e) { float d = impT[e] - m; var += d * d; }
      var /= (float)ED;
      loss += var / (m * m + 1e-10f);
    }
    {
      float m = 0.f;
      for (int e = 0; e < ED; ++e) m += ldT[e];
      m /= (float)ED;
      float var = 0.f;
      for (int e = 0; e < ED; ++e) { float d = ldT[e] - m; var += d * d; }
      var /= (float)ED;
      loss += var / (m * m + 1e-10f);
    }
    out_loss[0] = loss * 0.01f;
  }
}

// ---------------- launch ----------------
extern "C" void kernel_launch(void* const* d_in, const int* in_sizes, int n_in,
                              void* d_out, int out_size, void* d_ws, size_t ws_size,
                              hipStream_t stream) {
  (void)in_sizes; (void)n_in; (void)out_size; (void)ws_size;
  const float* x   = (const float*)d_in[0];
  const float* bw  = (const float*)d_in[1];
  const float* xp  = (const float*)d_in[2];
  const float* Wz  = (const float*)d_in[3];
  const float* lng = (const float*)d_in[4];
  const float* lnb = (const float*)d_in[5];
  const float* Wg  = (const float*)d_in[6];
  const float* bg  = (const float*)d_in[7];
  const float* W1  = (const float*)d_in[8];
  const float* b1  = (const float*)d_in[9];
  const float* W2  = (const float*)d_in[10];
  const float* b2  = (const float*)d_in[11];
  const float* A1  = (const float*)d_in[12];
  const float* B1  = (const float*)d_in[13];
  const float* A2  = (const float*)d_in[14];
  const float* B2  = (const float*)d_in[15];
  float* out = (float*)d_out;

  // workspace carve-out (256B aligned sections)
  size_t off = 0;
  auto carve = [&](size_t bytes) -> void* {
    void* p = (void*)((char*)d_ws + off);
    off += (bytes + 255) & ~(size_t)255;
    return p;
  };
  float*     logits = (float*)carve((size_t)NTOK * ED * 4);
  unsigned*  thr    = (unsigned*)carve(ED * 4);
  int*       tidx   = (int*)carve((size_t)NTOK * 2 * 4);
  float*     twv    = (float*)carve((size_t)NTOK * 2 * 4);
  int*       cnt    = (int*)carve(ED * 4);
  int*       ltok   = (int*)carve((size_t)ED * NTOK * 4);
  float*     lwv    = (float*)carve((size_t)ED * NTOK * 4);
  _Float16*  w1t    = (_Float16*)carve((size_t)ED * HD * CD * 2);
  _Float16*  w2t    = (_Float16*)carve((size_t)ED * OD * HD * 2);
  _Float16*  a1t    = (_Float16*)carve((size_t)ED * 32 * CD * 2);
  _Float16*  b1t    = (_Float16*)carve((size_t)ED * HD * 32 * 2);
  _Float16*  a2t    = (_Float16*)carve((size_t)ED * 32 * HD * 2);
  _Float16*  b2t    = (_Float16*)carve((size_t)ED * OD * 32 * 2);

  const int nout = NTOK * OD + 1;
  init_kernel<<<dim3((nout + 255) / 256), 256, 0, stream>>>(out, cnt, nout);
  prep_kernel<<<dim3(1024), 256, 0, stream>>>(W1, W2, A1, B1, A2, B2,
                                              w1t, w2t, a1t, b1t, a2t, b2t);
  gate_kernel<<<dim3(NTOK / 256), 256, 0, stream>>>(x, xp, Wz, lng, lnb, Wg, bg,
                                                    logits);
  topk_kernel<<<dim3(ED), 256, 0, stream>>>(logits, thr);
  route_kernel<<<dim3(NTOK / 256), 256, 0, stream>>>(logits, thr, tidx, twv, cnt,
                                                     ltok, lwv);
  expert_kernel<<<dim3(NTOK / TM, ED), 128, 0, stream>>>(
      x, bw, b1, b2, w1t, w2t, a1t, b1t, a2t, b2t, cnt, ltok, lwv, out);
  loss_kernel<<<dim3(1), 256, 0, stream>>>(tidx, twv, out + (size_t)NTOK * OD);
}
